// TransformerBlock_523986010183
// MI455X (gfx1250) — compile-verified, matrix operand order checked by reading, not run on hardware
//
#include <hip/hip_runtime.h>

// ---------------------------------------------------------------------------
// Transformer block forward for MI455X (gfx1250, wave32, WMMA).
// All matmuls on v_wmma_f32_16x16x32_bf16 (f32 accumulate).
// All tile staging is global_load_async_to_lds_b128; fragments are
// ds_load_b128 pairs (layout-aware). Weights pre-transposed once; V written
// transposed by its projection GEMM so flash staging is copy-only.
// ---------------------------------------------------------------------------

typedef __bf16 bf16;
typedef __attribute__((ext_vector_type(16))) __bf16 v16bf;
typedef __attribute__((ext_vector_type(8)))  __bf16 v8bf;
typedef __attribute__((ext_vector_type(8)))  float  v8f;
typedef int v4i __attribute__((vector_size(16)));

#define D_MODEL 1024
#define N_HEADS 16
#define D_K     64
#define D_FF    4096
#define SEQ     2048
#define BATCH   2
#define M_TOT   (BATCH * SEQ)   // 4096 tokens

// LDS row paddings (elements): keep 16B alignment, spread banks
#define PAD40 40   // for 32-wide k rows  (80 B stride)
#define PAD72 72   // for 64-wide d rows  (144 B stride)

// float -> bf16, round-to-nearest-even
static __device__ __forceinline__ bf16 f2bf(float f) {
    unsigned u = __builtin_bit_cast(unsigned, f);
    unsigned r = u + 0x7FFFu + ((u >> 16) & 1u);
    unsigned short h = (unsigned short)(r >> 16);
    return __builtin_bit_cast(bf16, h);
}

// Load a v16bf WMMA fragment whose K-dim is contiguous at `row` base:
// lane elements are rows[half*8 .. +8) and rows[16+half*8 .. +8).
static __device__ __forceinline__ v16bf load_frag(const bf16* row, int half) {
    v8bf lo = *(const v8bf*)(row + half * 8);
    v8bf hi = *(const v8bf*)(row + 16 + half * 8);
    return __builtin_shufflevector(lo, hi, 0, 1, 2, 3, 4, 5, 6, 7,
                                   8, 9, 10, 11, 12, 13, 14, 15);
}

// 16-byte global->LDS copy; gfx1250 async path (prototype takes v4i*).
static __device__ __forceinline__ void cp16_g2l(const bf16* g, bf16* l) {
#if __has_builtin(__builtin_amdgcn_global_load_async_to_lds_b128)
    __builtin_amdgcn_global_load_async_to_lds_b128((v4i*)g, (v4i*)l, 0, 0);
#else
    *(v8bf*)l = *(const v8bf*)g;
#endif
}
static __device__ __forceinline__ void wait_cp_async() {
#if __has_builtin(__builtin_amdgcn_global_load_async_to_lds_b128)
#if __has_builtin(__builtin_amdgcn_s_wait_asynccnt)
    __builtin_amdgcn_s_wait_asynccnt(0);
#else
    asm volatile("s_wait_asynccnt 0" ::: "memory");
#endif
#endif
}

// reductions across the 16-lane row group (wave32)
static __device__ __forceinline__ float rowmax16(float v) {
    #pragma unroll
    for (int m = 1; m < 16; m <<= 1) v = fmaxf(v, __shfl_xor(v, m, 32));
    return v;
}
static __device__ __forceinline__ float rowsum16(float v) {
    #pragma unroll
    for (int m = 1; m < 16; m <<= 1) v += __shfl_xor(v, m, 32);
    return v;
}

// ---------------------------------------------------------------------------
// f32 -> bf16 conversion (row-major copy)
// ---------------------------------------------------------------------------
__global__ void cvt_f32_bf16(const float* __restrict__ in, bf16* __restrict__ out, int n) {
    int i = blockIdx.x * blockDim.x + threadIdx.x;
    if (i < n) out[i] = f2bf(in[i]);
}

// ---------------------------------------------------------------------------
// f32 [K,N] -> bf16 [N,K] transpose (one-time, LDS-tiled 32x32)
// grid: (K/32, N/32), block 256
// ---------------------------------------------------------------------------
__global__ __launch_bounds__(256)
void cvt_transpose_f32_bf16(const float* __restrict__ in, bf16* __restrict__ out,
                            int K, int N) {
    __shared__ bf16 Ts[32][33];
    const int tid = threadIdx.x;
    const int tk = blockIdx.x * 32, tn = blockIdx.y * 32;
    #pragma unroll
    for (int it = 0; it < 4; ++it) {
        int i = tid + it * 256;
        int r = i >> 5, c = i & 31;                       // k-off, n-off
        Ts[r][c] = f2bf(in[(size_t)(tk + r) * N + tn + c]);
    }
    __syncthreads();
    #pragma unroll
    for (int it = 0; it < 4; ++it) {
        int i = tid + it * 256;
        int r = i >> 5, c = i & 31;                       // n-off, k-off
        out[(size_t)(tn + r) * K + tk + c] = Ts[c][r];
    }
}

// ---------------------------------------------------------------------------
// Dense GEMM: C[M,N] = A[M,K](bf16,row) x BT[N,K](bf16, pre-transposed) + bias
// 4 waves / block, 64x64 tile, each wave owns a 16x64 strip (4 WMMA accs).
// Both A and BT tiles staged with async b128 copies (K-contiguous rows).
// Optional ReLU; writes f32 C, bf16 C, and/or bf16 C-transposed (CbT[n][m]).
// ---------------------------------------------------------------------------
__global__ __launch_bounds__(128)
void gemm_bf16_wmma(const bf16* __restrict__ A, const bf16* __restrict__ BT,
                    const float* __restrict__ bias,
                    float* __restrict__ Cf, bf16* __restrict__ Cb,
                    bf16* __restrict__ CbT,
                    int M, int N, int K, int relu) {
    __shared__ __align__(16) bf16 As[64][PAD40];    // [m][k]
    __shared__ __align__(16) bf16 Bs[64][PAD40];    // [n][k]

    const int tid  = threadIdx.x;
    const int w    = tid >> 5;        // wave 0..3
    const int L    = tid & 31;        // lane
    const int m16  = L & 15;
    const int half = L >> 4;
    const int tm   = blockIdx.x * 64;
    const int tn   = blockIdx.y * 64;

    v8f acc0 = {}, acc1 = {}, acc2 = {}, acc3 = {};
    const int nkb = K >> 5;

    for (int kb = 0; kb < nkb; ++kb) {
        __syncthreads();
        // stage A and BT tiles, 256 async 16B chunks each
        #pragma unroll
        for (int it = 0; it < 2; ++it) {
            int i = tid + it * 128;               // chunk id 0..255
            int r = i >> 2, c0 = (i & 3) * 8;     // row, k base
            cp16_g2l(&A [(size_t)(tm + r) * K + kb * 32 + c0], &As[r][c0]);
            cp16_g2l(&BT[(size_t)(tn + r) * K + kb * 32 + c0], &Bs[r][c0]);
        }
        if (kb + 1 < nkb) {   // gfx1250 global_prefetch_b8 for next tiles
            __builtin_prefetch(&A [(size_t)(tm + (tid >> 1)) * K + (kb + 1) * 32], 0, 0);
            __builtin_prefetch(&BT[(size_t)(tn + (tid >> 1)) * K + (kb + 1) * 32], 0, 0);
        }
        wait_cp_async();
        __syncthreads();

        v16bf a  = load_frag(&As[w * 16 + m16][0], half);
        v16bf b0 = load_frag(&Bs[ 0 + m16][0], half);
        v16bf b1 = load_frag(&Bs[16 + m16][0], half);
        v16bf b2 = load_frag(&Bs[32 + m16][0], half);
        v16bf b3 = load_frag(&Bs[48 + m16][0], half);

        acc0 = __builtin_amdgcn_wmma_f32_16x16x32_bf16(false, a, false, b0, (short)0, acc0, false, false);
        acc1 = __builtin_amdgcn_wmma_f32_16x16x32_bf16(false, a, false, b1, (short)0, acc1, false, false);
        acc2 = __builtin_amdgcn_wmma_f32_16x16x32_bf16(false, a, false, b2, (short)0, acc2, false, false);
        acc3 = __builtin_amdgcn_wmma_f32_16x16x32_bf16(false, a, false, b3, (short)0, acc3, false, false);
    }

    // epilogue: bias (+relu); f32 / bf16 / bf16-transposed stores
    #pragma unroll
    for (int r = 0; r < 8; ++r) {
        int row = tm + w * 16 + r + 8 * half;
        #pragma unroll
        for (int j = 0; j < 4; ++j) {
            int col = tn + j * 16 + m16;
            float v = (j == 0 ? acc0[r] : j == 1 ? acc1[r] : j == 2 ? acc2[r] : acc3[r]);
            if (bias) v += bias[col];
            if (relu) v = fmaxf(v, 0.0f);
            if (Cf)  Cf [(size_t)row * N + col] = v;
            if (Cb)  Cb [(size_t)row * N + col] = f2bf(v);
            if (CbT) CbT[(size_t)col * M + row] = f2bf(v);
        }
    }
}

// ---------------------------------------------------------------------------
// Flash attention. Grid: (S/64, H, B), 4 waves/blk; each wave one 16-row
// query tile; loop over 64 key blocks of 32 keys.
// K staged row-major [key][d]; V comes PRE-TRANSPOSED (VT[col][token]) so its
// [d][key] tile is a straight async copy. P transposed via per-wave LDS.
// ---------------------------------------------------------------------------
__global__ __launch_bounds__(128)
void flash_attn_wmma(const bf16* __restrict__ Q, const bf16* __restrict__ Kg,
                     const bf16* __restrict__ VT, bf16* __restrict__ O) {
    __shared__ __align__(16) bf16 Ks[32][PAD72];       // [key][d]
    __shared__ __align__(16) bf16 Vs[64][PAD40];       // [d][key]
    __shared__ __align__(16) bf16 Ps[4][16][PAD40];    // per-wave [qrow][key]

    const int tid  = threadIdx.x;
    const int w    = tid >> 5;
    const int L    = tid & 31;
    const int m16  = L & 15;
    const int half = L >> 4;
    const int qb = blockIdx.x, h = blockIdx.y, b = blockIdx.z;
    const int qtok0 = b * SEQ + qb * 64 + w * 16;
    const int cbase = h * D_K;

    // Q fragments (16x64 -> two 16x32 A-frags) straight from global (b128 x4)
    const bf16* qrow = &Q[(size_t)(qtok0 + m16) * D_MODEL + cbase];
    v16bf qa0 = load_frag(qrow, half);
    v16bf qa1 = load_frag(qrow + 32, half);

    float mrow[8], lrow[8];
    #pragma unroll
    for (int r = 0; r < 8; ++r) { mrow[r] = -1e30f; lrow[r] = 0.0f; }
    v8f o0 = {}, o1 = {}, o2 = {}, o3 = {};

    const int ktok0 = b * SEQ;
    for (int kb = 0; kb < SEQ / 32; ++kb) {
        __syncthreads();
        // K block [32 keys x 64 d]: 256 async 16B chunks
        #pragma unroll
        for (int it = 0; it < 2; ++it) {
            int i = tid + it * 128;
            int r = i >> 3, c0 = (i & 7) * 8;   // key row, d base
            cp16_g2l(&Kg[(size_t)(ktok0 + kb * 32 + r) * D_MODEL + cbase + c0], &Ks[r][c0]);
        }
        // V block [64 d x 32 keys] from VT[col][token]: 256 async 16B chunks
        #pragma unroll
        for (int it = 0; it < 2; ++it) {
            int i = tid + it * 128;
            int r = i >> 2, c0 = (i & 3) * 8;   // d row, key base
            cp16_g2l(&VT[(size_t)(cbase + r) * M_TOT + ktok0 + kb * 32 + c0], &Vs[r][c0]);
        }
        wait_cp_async();
        __syncthreads();

        // scores for two 16-key subtiles: s[t] = Q(16x64) . K_t(16x64)^T
        v8f s[2];
        #pragma unroll
        for (int t = 0; t < 2; ++t) {
            v16bf bb0 = load_frag(&Ks[t * 16 + m16][0],  half);   // d 0..31
            v16bf bb1 = load_frag(&Ks[t * 16 + m16][32], half);   // d 32..63
            v8f c = {};
            c = __builtin_amdgcn_wmma_f32_16x16x32_bf16(false, qa0, false, bb0, (short)0, c, false, false);
            c = __builtin_amdgcn_wmma_f32_16x16x32_bf16(false, qa1, false, bb1, (short)0, c, false, false);
            s[t] = c;
        }

        // online softmax update over this 32-key block
        #pragma unroll
        for (int r = 0; r < 8; ++r) {
            float a0 = s[0][r] * 0.125f;   // 1/sqrt(64)
            float a1 = s[1][r] * 0.125f;
            float bm = rowmax16(fmaxf(a0, a1));
            float nm = fmaxf(mrow[r], bm);
            float sc = __expf(mrow[r] - nm);
            float p0 = __expf(a0 - nm);
            float p1 = __expf(a1 - nm);
            lrow[r] = lrow[r] * sc + rowsum16(p0 + p1);
            mrow[r] = nm;
            o0[r] *= sc; o1[r] *= sc; o2[r] *= sc; o3[r] *= sc;
            int row = r + 8 * half;
            Ps[w][row][m16]      = f2bf(p0);   // C-layout -> [row][key] in LDS
            Ps[w][row][16 + m16] = f2bf(p1);
        }
        __syncthreads();

        // P (16x32) as A-frag; V fragments; O += P.V
        v16bf pa  = load_frag(&Ps[w][m16][0], half);
        v16bf vb0 = load_frag(&Vs[ 0 + m16][0], half);
        v16bf vb1 = load_frag(&Vs[16 + m16][0], half);
        v16bf vb2 = load_frag(&Vs[32 + m16][0], half);
        v16bf vb3 = load_frag(&Vs[48 + m16][0], half);
        o0 = __builtin_amdgcn_wmma_f32_16x16x32_bf16(false, pa, false, vb0, (short)0, o0, false, false);
        o1 = __builtin_amdgcn_wmma_f32_16x16x32_bf16(false, pa, false, vb1, (short)0, o1, false, false);
        o2 = __builtin_amdgcn_wmma_f32_16x16x32_bf16(false, pa, false, vb2, (short)0, o2, false, false);
        o3 = __builtin_amdgcn_wmma_f32_16x16x32_bf16(false, pa, false, vb3, (short)0, o3, false, false);
    }

    // normalize and store bf16 attention output [token, h*64 + d]
    #pragma unroll
    for (int r = 0; r < 8; ++r) {
        int row = r + 8 * half;
        float inv = 1.0f / lrow[r];
        int tok = qtok0 + row;
        O[(size_t)tok * D_MODEL + cbase +  0 + m16] = f2bf(o0[r] * inv);
        O[(size_t)tok * D_MODEL + cbase + 16 + m16] = f2bf(o1[r] * inv);
        O[(size_t)tok * D_MODEL + cbase + 32 + m16] = f2bf(o2[r] * inv);
        O[(size_t)tok * D_MODEL + cbase + 48 + m16] = f2bf(o3[r] * inv);
    }
}

// ---------------------------------------------------------------------------
// Fused residual-add + LayerNorm over D_MODEL=1024. One block (256 thr)/row.
// ---------------------------------------------------------------------------
__global__ __launch_bounds__(256)
void add_layernorm(const float* __restrict__ A, const float* __restrict__ Bv,
                   const float* __restrict__ g, const float* __restrict__ be,
                   float* __restrict__ outf, bf16* __restrict__ outb) {
    __shared__ float red[256];
    const int row = blockIdx.x, tid = threadIdx.x;

    float v[4];
    float s = 0.0f;
    #pragma unroll
    for (int i = 0; i < 4; ++i) {
        int c = tid + i * 256;
        v[i] = A[(size_t)row * D_MODEL + c] + Bv[(size_t)row * D_MODEL + c];
        s += v[i];
    }
    red[tid] = s; __syncthreads();
    for (int st = 128; st > 0; st >>= 1) { if (tid < st) red[tid] += red[tid + st]; __syncthreads(); }
    float mean = red[0] * (1.0f / D_MODEL);
    __syncthreads();

    float s2 = 0.0f;
    #pragma unroll
    for (int i = 0; i < 4; ++i) { float d = v[i] - mean; s2 += d * d; }
    red[tid] = s2; __syncthreads();
    for (int st = 128; st > 0; st >>= 1) { if (tid < st) red[tid] += red[tid + st]; __syncthreads(); }
    float rstd = rsqrtf(red[0] * (1.0f / D_MODEL) + 1e-5f);

    #pragma unroll
    for (int i = 0; i < 4; ++i) {
        int c = tid + i * 256;
        float o = (v[i] - mean) * rstd * g[c] + be[c];
        if (outf) outf[(size_t)row * D_MODEL + c] = o;
        if (outb) outb[(size_t)row * D_MODEL + c] = f2bf(o);
    }
}

// ---------------------------------------------------------------------------
// Host launcher
// ---------------------------------------------------------------------------
extern "C" void kernel_launch(void* const* d_in, const int* in_sizes, int n_in,
                              void* d_out, int out_size, void* d_ws, size_t ws_size,
                              hipStream_t stream) {
    const float* x    = (const float*)d_in[0];
    const float* wq   = (const float*)d_in[1];
    const float* bq   = (const float*)d_in[2];
    const float* wk   = (const float*)d_in[3];
    const float* bk   = (const float*)d_in[4];
    const float* wv   = (const float*)d_in[5];
    const float* bv   = (const float*)d_in[6];
    const float* wo   = (const float*)d_in[7];
    const float* bo   = (const float*)d_in[8];
    const float* ln1g = (const float*)d_in[9];
    const float* ln1b = (const float*)d_in[10];
    const float* ln2g = (const float*)d_in[11];
    const float* ln2b = (const float*)d_in[12];
    const float* w1   = (const float*)d_in[13];
    const float* b1   = (const float*)d_in[14];
    const float* w2   = (const float*)d_in[15];
    const float* b2   = (const float*)d_in[16];

    // ---- carve workspace ----
    char* p = (char*)d_ws;
    auto carve = [&](size_t bytes) -> char* {
        char* q = p;
        p += (bytes + 255) & ~(size_t)255;
        return q;
    };
    const size_t MD = (size_t)M_TOT * D_MODEL;
    const size_t DD = (size_t)D_MODEL * D_MODEL;
    const size_t DF = (size_t)D_MODEL * D_FF;
    const size_t MF = (size_t)M_TOT * D_FF;

    bf16* x_bf   = (bf16*)carve(MD * 2);
    bf16* wqT    = (bf16*)carve(DD * 2);   // [out,in] transposed weights
    bf16* wkT    = (bf16*)carve(DD * 2);
    bf16* wvT    = (bf16*)carve(DD * 2);
    bf16* woT    = (bf16*)carve(DD * 2);
    bf16* w1T    = (bf16*)carve(DF * 2);
    bf16* w2T    = (bf16*)carve(DF * 2);
    bf16* q_bf   = (bf16*)carve(MD * 2);
    bf16* k_bf   = (bf16*)carve(MD * 2);
    bf16* vT_bf  = (bf16*)carve(MD * 2);   // V transposed: [col][token]
    bf16* at_bf  = (bf16*)carve(MD * 2);
    float* tmp1  = (float*)carve(MD * 4);
    float* x1_f  = (float*)carve(MD * 4);
    bf16* x1_bf  = (bf16*)carve(MD * 2);
    bf16* h_bf   = (bf16*)carve(MF * 2);
    float* tmp2  = (float*)carve(MD * 4);

    // 1) convert activations; convert+transpose all weights (one-time)
    cvt_f32_bf16<<<(unsigned)((MD + 255) / 256), 256, 0, stream>>>(x, x_bf, (int)MD);
    dim3 t256(256);
    cvt_transpose_f32_bf16<<<dim3(D_MODEL / 32, D_MODEL / 32), t256, 0, stream>>>(wq, wqT, D_MODEL, D_MODEL);
    cvt_transpose_f32_bf16<<<dim3(D_MODEL / 32, D_MODEL / 32), t256, 0, stream>>>(wk, wkT, D_MODEL, D_MODEL);
    cvt_transpose_f32_bf16<<<dim3(D_MODEL / 32, D_MODEL / 32), t256, 0, stream>>>(wv, wvT, D_MODEL, D_MODEL);
    cvt_transpose_f32_bf16<<<dim3(D_MODEL / 32, D_MODEL / 32), t256, 0, stream>>>(wo, woT, D_MODEL, D_MODEL);
    cvt_transpose_f32_bf16<<<dim3(D_MODEL / 32, D_FF / 32),    t256, 0, stream>>>(w1, w1T, D_MODEL, D_FF);
    cvt_transpose_f32_bf16<<<dim3(D_FF / 32, D_MODEL / 32),    t256, 0, stream>>>(w2, w2T, D_FF, D_MODEL);

    dim3 blk(128);
    dim3 gDD(M_TOT / 64, D_MODEL / 64);  // (64,16)
    dim3 gDF(M_TOT / 64, D_FF / 64);     // (64,64)

    // 2) QKV projections (Q,K row-major bf16; V transposed bf16)
    gemm_bf16_wmma<<<gDD, blk, 0, stream>>>(x_bf, wqT, bq, nullptr, q_bf, nullptr, M_TOT, D_MODEL, D_MODEL, 0);
    gemm_bf16_wmma<<<gDD, blk, 0, stream>>>(x_bf, wkT, bk, nullptr, k_bf, nullptr, M_TOT, D_MODEL, D_MODEL, 0);
    gemm_bf16_wmma<<<gDD, blk, 0, stream>>>(x_bf, wvT, bv, nullptr, nullptr, vT_bf, M_TOT, D_MODEL, D_MODEL, 0);

    // 3) flash attention -> bf16 [token, h*64+d]
    flash_attn_wmma<<<dim3(SEQ / 64, N_HEADS, BATCH), blk, 0, stream>>>(q_bf, k_bf, vT_bf, at_bf);

    // 4) output projection (f32)
    gemm_bf16_wmma<<<gDD, blk, 0, stream>>>(at_bf, woT, bo, tmp1, nullptr, nullptr, M_TOT, D_MODEL, D_MODEL, 0);

    // 5) residual + LN1 -> x1 (f32 + bf16)
    add_layernorm<<<M_TOT, 256, 0, stream>>>(x, tmp1, ln1g, ln1b, x1_f, x1_bf);

    // 6) FFN up (ReLU fused, bf16 out)
    gemm_bf16_wmma<<<gDF, blk, 0, stream>>>(x1_bf, w1T, b1, nullptr, h_bf, nullptr, M_TOT, D_FF, D_MODEL, 1);

    // 7) FFN down (f32)
    gemm_bf16_wmma<<<gDD, blk, 0, stream>>>(h_bf, w2T, b2, tmp2, nullptr, nullptr, M_TOT, D_MODEL, D_FF, 0);

    // 8) residual + LN2 -> final f32 output
    add_layernorm<<<M_TOT, 256, 0, stream>>>(x1_f, tmp2, ln2g, ln2b, (float*)d_out, nullptr);
}
